// FrameTransformerEncoder_47588237640238
// MI455X (gfx1250) — matrix operand order repaired and checked by reading, not compiled, not run on hardware
//
#include <hip/hip_runtime.h>

// ---------------------------------------------------------------------------
// FrameTransformerEncoder for MI455X (gfx1250, wave32, WMMA bf16 + TDM)
// ---------------------------------------------------------------------------

constexpr int cB = 2, cC = 8, cF = 512, cW = 1024, cNH = 8, cHD = 64, cEXP = 4;
constexpr float cEPS = 1e-5f;
constexpr int NX = cB * cC * cF * cW;          // 8388608 elements
constexpr int GK = cF;                          // GEMM K (512)
constexpr int GN = cW;                          // GEMM N (1024)
constexpr float ATT_SCALE = 0.04419417382415922f; // 1/sqrt(F=512)

typedef __bf16 v8bf  __attribute__((ext_vector_type(8)));
typedef __bf16 v16bf __attribute__((ext_vector_type(16)));
typedef float  v8f   __attribute__((ext_vector_type(8)));
typedef unsigned int v4u __attribute__((ext_vector_type(4)));
typedef int v4i __attribute__((ext_vector_type(4)));
typedef int v8i __attribute__((ext_vector_type(8)));

#define SHUF16(lo, hi) \
  __builtin_shufflevector((lo), (hi), 0,1,2,3,4,5,6,7,8,9,10,11,12,13,14,15)
#define WMMA_BF16(a, b, c) \
  __builtin_amdgcn_wmma_f32_16x16x32_bf16(false, (a), false, (b), (short)0, (c), false, false)

#if defined(__AMDGCN__) && __has_builtin(__builtin_amdgcn_tensor_load_to_lds) && \
    __has_builtin(__builtin_amdgcn_s_wait_tensorcnt)
#define HAVE_TDM 1
#else
#define HAVE_TDM 0
#endif

__device__ __forceinline__ float sqrelu(float x) { float r = fmaxf(x, 0.f); return r * r; }
__device__ __forceinline__ float sigmoidf(float x) { return 1.f / (1.f + __expf(-x)); }

#if HAVE_TDM
// Issue one TENSOR_LOAD_TO_LDS for a 2D bf16 tile:
//   tile_rows rows of tile_k elements, row stride = tensor_k elements,
//   packed row-major into LDS at lds_dst. Tracked with TENSORcnt.
__device__ __forceinline__ void tdm_load_bf16_2d(const __bf16* gsrc, __bf16* lds_dst,
                                                 int tile_k, int tile_rows,
                                                 int tensor_k, int tensor_rows) {
  unsigned long long ga = (unsigned long long)gsrc;
  unsigned int lds = (unsigned int)(unsigned long long)lds_dst; // low 32 bits = LDS offset
  // ---- D# group 0 (ISA 8.3): count=1 | lds_addr | global_addr | type=2 ----
  v4u g0;
  g0.x = 1u;                                            // count=1, user descriptor
  g0.y = lds;                                           // bits 63:32
  g0.z = (unsigned int)(ga & 0xffffffffull);            // global_addr[31:0]
  g0.w = (unsigned int)((ga >> 32) & 0x01ffffffull) |   // global_addr[56:32]
         (2u << 30);                                    // type = 2 ("image")
  // ---- D# group 1 (ISA 8.4) ----
  unsigned int td0 = (unsigned int)tensor_k;
  unsigned int td1 = (unsigned int)tensor_rows;
  unsigned int st0 = (unsigned int)tensor_k;            // dim0 stride (elements)
  v8i g1;
  g1[0] = (int)(1u << 16);                              // wg_mask=0, data_size=1 (2B)
  g1[1] = (int)((td0 & 0xffffu) << 16);                 // tensor_dim0[15:0]
  g1[2] = (int)(((td0 >> 16) & 0xffffu) | ((td1 & 0xffffu) << 16));
  g1[3] = (int)(((td1 >> 16) & 0xffffu) |
                (((unsigned int)tile_k & 0xffffu) << 16)); // tile_dim0
  g1[4] = (int)((unsigned int)tile_rows & 0xffffu);     // tile_dim1 (tile_dim2=0)
  g1[5] = (int)st0;                                     // dim0_stride[31:0]
  g1[6] = 0;                                            // dim0_stride[47:32], dim1_stride lo
  g1[7] = 0;
  v4i z4 = {0, 0, 0, 0};
#if __clang_major__ >= 23
  v8i z8 = {0, 0, 0, 0, 0, 0, 0, 0};
  __builtin_amdgcn_tensor_load_to_lds(g0, g1, z4, z4, z8, 0);
#else
  __builtin_amdgcn_tensor_load_to_lds(g0, g1, z4, z4, 0);
#endif
}
#endif // HAVE_TDM

// --------------------------- float -> bf16 convert --------------------------
__global__ void k_cvt_bf16(const float* __restrict__ in, __bf16* __restrict__ out, int n) {
  int i = blockIdx.x * blockDim.x + threadIdx.x;
  if (i < n) out[i] = (__bf16)in[i];
}

// --------------------------- per-channel layernorm ---------------------------
// grid: (B*C, W/16), block 256. f32 out natural [bc][f][w]; bf16 out w-major
// [bc][w][f] so GEMM B-tiles are contiguous row copies (TDM-friendly).
__global__ __launch_bounds__(256) void k_layernorm(
    const float* __restrict__ X, const float* __restrict__ G, const float* __restrict__ Bt,
    float* __restrict__ Of, __bf16* __restrict__ Ob) {
  int bc = blockIdx.x;
  int c  = bc % cC;
  int w0 = blockIdx.y * 16;
  int t  = threadIdx.x;
  int wc = t & 15, ig = t >> 4;
  const float* xp = X + (size_t)bc * cF * cW + w0 + wc;
  float vbuf[32];
  float s = 0.f, q = 0.f;
#pragma unroll
  for (int j = 0; j < 32; ++j) {
    float val = xp[(size_t)(ig + 16 * j) * cW];
    vbuf[j] = val; s += val; q += val * val;
  }
  __shared__ float shs[256], shq[256];
  shs[t] = s; shq[t] = q;
  __syncthreads();
#pragma unroll
  for (int st = 128; st >= 16; st >>= 1) {
    if (t < st) { shs[t] += shs[t + st]; shq[t] += shq[t + st]; }
    __syncthreads();
  }
  float mu  = shs[wc] * (1.f / cF);
  float var = shq[wc] * (1.f / cF) - mu * mu;
  float rs  = rsqrtf(var + cEPS);
  float*  ofp = Of + (size_t)bc * cF * cW + w0 + wc;
  __bf16* obp = Ob + ((size_t)bc * cW + (w0 + wc)) * cF;   // transposed row
#pragma unroll
  for (int j = 0; j < 32; ++j) {
    int i = ig + 16 * j;
    float y = (vbuf[j] - mu) * rs * G[c * cF + i] + Bt[c * cF + i];
    ofp[(size_t)i * cW] = y;
    obp[i] = (__bf16)y;
  }
}

// --------------------------- WMMA bf16 GEMM ---------------------------------
// out[bc][m][n] (+)= sum_k A[c][m][k] * Bt[bc][n][k]   (B stored w-major!)
// grid: (N/128, M/64, B*C); block 256 (8 waves). Double-buffered LDS staging:
// TDM tensor_load_to_lds when available, contiguous b128 copies otherwise.
__global__ __launch_bounds__(256) void k_gemm(
    const __bf16* __restrict__ A, const __bf16* __restrict__ Bm,
    float* __restrict__ Out, int M, int accum) {
  int bc = blockIdx.z, c = bc % cC;
  int n0 = blockIdx.x * 128;
  int m0 = blockIdx.y * 64;
  const __bf16* Ap = A  + (size_t)c  * M  * GK;
  const __bf16* Bp = Bm + (size_t)bc * GN * GK;     // [n][k]
  __shared__ __align__(16) __bf16 As[2][64 * 32];
  __shared__ __align__(16) __bf16 Bs[2][128 * 32];
  int t = threadIdx.x;
  int wv = t >> 5, lane = t & 31, half = lane >> 4, lr = lane & 15;
  int wm = wv & 3, wn = wv >> 2;

  v8f acc[4];
#pragma unroll
  for (int nt = 0; nt < 4; ++nt)
#pragma unroll
    for (int e = 0; e < 8; ++e) acc[nt][e] = 0.f;

  constexpr int KT = GK / 32;   // 16 K-tiles

  auto compute = [&](const __bf16* Asb, const __bf16* Bsb) {
    int arow = wm * 16 + lr;
    v8bf alo = *(const v8bf*)&Asb[arow * 32 + half * 8];
    v8bf ahi = *(const v8bf*)&Asb[arow * 32 + 16 + half * 8];
    v16bf af = SHUF16(alo, ahi);
#pragma unroll
    for (int nt = 0; nt < 4; ++nt) {
      int bcol = wn * 64 + nt * 16 + lr;
      v8bf blo = *(const v8bf*)&Bsb[bcol * 32 + half * 8];
      v8bf bhi = *(const v8bf*)&Bsb[bcol * 32 + 16 + half * 8];
      v16bf bf = SHUF16(blo, bhi);
      acc[nt] = WMMA_BF16(af, bf, acc[nt]);
    }
  };

#if HAVE_TDM
  // ---- Tensor Data Mover path: wave 0 drives DMA, all waves compute ----
  if (wv == 0) {
    tdm_load_bf16_2d(Ap + (size_t)m0 * GK, As[0], 32, 64, GK, M);
    tdm_load_bf16_2d(Bp + (size_t)n0 * GK, Bs[0], 32, 128, GK, GN);
    __builtin_amdgcn_s_wait_tensorcnt(0);
  }
  __syncthreads();
  for (int i = 0; i < KT; ++i) {
    int k0 = i * 32;
    int cur = i & 1, nxt = cur ^ 1;
    if (k0 + 32 < GK) {
      if (wv == 0) {
        tdm_load_bf16_2d(Ap + (size_t)m0 * GK + (k0 + 32), As[nxt], 32, 64, GK, M);
        tdm_load_bf16_2d(Bp + (size_t)n0 * GK + (k0 + 32), Bs[nxt], 32, 128, GK, GN);
      }
      // warm L2 for the tile after next (gfx1250 global_prefetch_b8)
      __builtin_prefetch((const void*)&Ap[(size_t)(m0 + (t >> 2)) * GK + k0 + 64], 0, 1);
      __builtin_prefetch((const void*)&Bp[(size_t)(n0 + (t >> 1)) * GK + k0 + 64], 0, 1);
    }
    compute(As[cur], Bs[cur]);
    if (wv == 0) __builtin_amdgcn_s_wait_tensorcnt(0);
    __syncthreads();
  }
#else
  // ---- fallback: cooperative contiguous b128 copies, software pipelined ----
  int am = t >> 2, ako = (t & 3) * 8;                 // A: 64 rows x 32 k
  int bn = t >> 2, bko = (t & 3) * 8;                 // B: 128 rows x 32 k (2 halves)
  *(v8bf*)&As[0][am * 32 + ako] = *(const v8bf*)&Ap[(size_t)(m0 + am) * GK + ako];
  *(v8bf*)&Bs[0][bn * 32 + bko] = *(const v8bf*)&Bp[(size_t)(n0 + bn) * GK + bko];
  *(v8bf*)&Bs[0][(64 + bn) * 32 + bko] =
      *(const v8bf*)&Bp[(size_t)(n0 + 64 + bn) * GK + bko];
  __syncthreads();
  for (int i = 0; i < KT; ++i) {
    int cur = i & 1, nxt = cur ^ 1;
    bool more = (i + 1 < KT);
    int k1 = (i + 1) * 32;
    v8bf ta, tb0, tb1;
    if (more) {
      ta  = *(const v8bf*)&Ap[(size_t)(m0 + am) * GK + k1 + ako];
      tb0 = *(const v8bf*)&Bp[(size_t)(n0 + bn) * GK + k1 + bko];
      tb1 = *(const v8bf*)&Bp[(size_t)(n0 + 64 + bn) * GK + k1 + bko];
      __builtin_prefetch((const void*)&Ap[(size_t)(m0 + am) * GK + k1 + 32 + ako], 0, 1);
      __builtin_prefetch((const void*)&Bp[(size_t)(n0 + bn) * GK + k1 + 32 + bko], 0, 1);
    }
    compute(As[cur], Bs[cur]);
    if (more) {
      *(v8bf*)&As[nxt][am * 32 + ako] = ta;
      *(v8bf*)&Bs[nxt][bn * 32 + bko] = tb0;
      *(v8bf*)&Bs[nxt][(64 + bn) * 32 + bko] = tb1;
    }
    __syncthreads();
  }
#endif

  float* Op = Out + (size_t)bc * M * GN;
  if (accum) {
#pragma unroll
    for (int nt = 0; nt < 4; ++nt)
#pragma unroll
      for (int r = 0; r < 8; ++r) {
        int row = m0 + wm * 16 + half * 8 + r;
        int col = n0 + wn * 64 + nt * 16 + lr;
        Op[(size_t)row * GN + col] += acc[nt][r];
      }
  } else {
#pragma unroll
    for (int nt = 0; nt < 4; ++nt)
#pragma unroll
      for (int r = 0; r < 8; ++r) {
        int row = m0 + wm * 16 + half * 8 + r;
        int col = n0 + wn * 64 + nt * 16 + lr;
        Op[(size_t)row * GN + col] = acc[nt][r];
      }
  }
}

// --------------------------- elementwise blocks ------------------------------
__global__ void k_glu_add(const float* __restrict__ Xin, const float* __restrict__ Z,
                          float* __restrict__ H) {
  size_t idx = (size_t)blockIdx.x * blockDim.x + threadIdx.x;
  int w = idx % cW;
  int f = (idx / cW) % cF;
  size_t bc = idx / ((size_t)cF * cW);
  size_t zb = bc * (size_t)(2 * cF) * cW;
  float a = Z[zb + (size_t)f * cW + w];
  float g = Z[zb + (size_t)(cF + f) * cW + w];
  H[idx] = Xin[idx] + a * sigmoidf(g);
}

__global__ void k_sqrelu_dw3_add(const float* __restrict__ Y1, const float* __restrict__ Z,
                                 const float* __restrict__ W1, const float* __restrict__ B1,
                                 float* __restrict__ H) {
  size_t idx = (size_t)blockIdx.x * blockDim.x + threadIdx.x;
  int f = (idx / cW) % cF;
  int c = (int)((idx / ((size_t)cF * cW)) % cC);
  float acc = B1[c];
#pragma unroll
  for (int j = 0; j < 3; ++j) {
    int ff = f - 1 + j;
    if (ff >= 0 && ff < cF) acc += W1[c * 3 + j] * Z[idx + (size_t)(ff - f) * cW];
  }
  H[idx] += sqrelu(Y1[idx]) + sqrelu(acc);
}

__global__ void k_dw9_add(const float* __restrict__ Z, const float* __restrict__ W2,
                          const float* __restrict__ B2, float* __restrict__ H) {
  size_t idx = (size_t)blockIdx.x * blockDim.x + threadIdx.x;
  int f = (idx / cW) % cF;
  int c = (int)((idx / ((size_t)cF * cW)) % cC);
  float acc = B2[c];
#pragma unroll
  for (int j = 0; j < 9; ++j) {
    int ff = f - 4 + j;
    if (ff >= 0 && ff < cF) acc += W2[c * 9 + j] * Z[idx + (size_t)(ff - f) * cW];
  }
  H[idx] += acc;
}

// frame conv (channel-mixing, kernel 3 along W, pad 1) + optional RoPE.
__global__ void k_convframes(const float* __restrict__ X, const float* __restrict__ CWt,
                             const float* __restrict__ CBv, __bf16* __restrict__ Out,
                             int rope, int transposed) {
  size_t idx = (size_t)blockIdx.x * blockDim.x + threadIdx.x;
  int w = idx % cW;
  size_t r = idx / cW;
  int f2 = (int)(r % (cF / 2)); r /= (cF / 2);
  int co = (int)(r % cC);
  int b  = (int)(r / cC);
  int f0 = f2 * 2;
  float a0 = CBv[co], a1 = CBv[co];
  for (int ci = 0; ci < cC; ++ci) {
    const float* xp = X + (((size_t)(b * cC + ci)) * cF + f0) * cW;
    const float* wr = CWt + (co * cC + ci) * 3;
#pragma unroll
    for (int j = 0; j < 3; ++j) {
      int ww = w - 1 + j;
      if (ww >= 0 && ww < cW) {
        a0 += wr[j] * xp[ww];
        a1 += wr[j] * xp[(size_t)cW + ww];
      }
    }
  }
  float o0 = a0, o1 = a1;
  if (rope) {
    int de = f0 % cHD;
    float inv = powf(10000.f, -(float)de / (float)cHD);
    float th = (float)w * inv;
    float cs = __cosf(th), sn = __sinf(th);
    o0 = a0 * cs - a1 * sn;
    o1 = a1 * cs + a0 * sn;
  }
  if (transposed) {  // head-major (b,c,h,w,d) for Q/K fragment loads
    int hh = f0 / cHD, de = f0 % cHD;
    size_t base = ((((size_t)(b * cC + co)) * cNH + hh) * cW + w) * cHD + de;
    Out[base] = (__bf16)o0;
    Out[base + 1] = (__bf16)o1;
  } else {           // natural (b,c,f,w) == per-head (d,w): V^T for P@V
    size_t base = (((size_t)(b * cC + co)) * cF + f0) * cW + w;
    Out[base] = (__bf16)o0;
    Out[base + cW] = (__bf16)o1;
  }
}

// --------------------------- flash attention (WMMA) --------------------------
// Q,K head-major [BC][NH][W][HD] bf16; V natural [BC][F][W] bf16.
// Output written w-major [BC][W][F] bf16 (GEMM B layout for the o-projection).
__global__ __launch_bounds__(256) void k_flash(
    const __bf16* __restrict__ Q, const __bf16* __restrict__ Kt,
    const __bf16* __restrict__ V, __bf16* __restrict__ O) {
  int bc = blockIdx.z, hh = blockIdx.y;
  int wv = threadIdx.x >> 5, lane = threadIdx.x & 31;
  int half = lane >> 4, lr = lane & 15;
  int q0 = blockIdx.x * 128 + wv * 16;
  const __bf16* qb = Q  + ((size_t)bc * cNH + hh) * cW * cHD;
  const __bf16* kb = Kt + ((size_t)bc * cNH + hh) * cW * cHD;
  const __bf16* vb = V  + ((size_t)bc * cF + hh * cHD) * cW;
  __shared__ __align__(16) __bf16 Pl[8][16 * 32];
  __bf16* pl = Pl[wv];

  v16bf qa[2];
#pragma unroll
  for (int ks = 0; ks < 2; ++ks) {
    const __bf16* qp = qb + (size_t)(q0 + lr) * cHD + ks * 32;
    v8bf lo = *(const v8bf*)(qp + half * 8);
    v8bf hi = *(const v8bf*)(qp + 16 + half * 8);
    qa[ks] = SHUF16(lo, hi);
  }
  float rm[8], l[8];
  v8f oacc[4];
#pragma unroll
  for (int r = 0; r < 8; ++r) { rm[r] = -1e30f; l[r] = 0.f; }
#pragma unroll
  for (int nt = 0; nt < 4; ++nt)
#pragma unroll
    for (int r = 0; r < 8; ++r) oacc[nt][r] = 0.f;

  for (int kv = 0; kv < cW; kv += 32) {
    v8f s0, s1;
#pragma unroll
    for (int e = 0; e < 8; ++e) { s0[e] = 0.f; s1[e] = 0.f; }
#pragma unroll
    for (int ks = 0; ks < 2; ++ks) {
      const __bf16* kp0 = kb + (size_t)(kv + lr) * cHD + ks * 32;
      v8bf l0 = *(const v8bf*)(kp0 + half * 8);
      v8bf h0 = *(const v8bf*)(kp0 + 16 + half * 8);
      s0 = WMMA_BF16(qa[ks], SHUF16(l0, h0), s0);
      const __bf16* kp1 = kp0 + (size_t)16 * cHD;
      v8bf l1 = *(const v8bf*)(kp1 + half * 8);
      v8bf h1 = *(const v8bf*)(kp1 + 16 + half * 8);
      s1 = WMMA_BF16(qa[ks], SHUF16(l1, h1), s1);
    }
    // online softmax; rows are (half*8 + r), columns striped across 16 lanes
#pragma unroll
    for (int r = 0; r < 8; ++r) {
      float e0 = s0[r] * ATT_SCALE, e1 = s1[r] * ATT_SCALE;
      float tm = fmaxf(e0, e1);
      tm = fmaxf(tm, __shfl_xor(tm, 1));
      tm = fmaxf(tm, __shfl_xor(tm, 2));
      tm = fmaxf(tm, __shfl_xor(tm, 4));
      tm = fmaxf(tm, __shfl_xor(tm, 8));
      float nm = fmaxf(rm[r], tm);
      float corr = __expf(rm[r] - nm);
      rm[r] = nm;
      float p0 = __expf(e0 - nm), p1 = __expf(e1 - nm);
      float rs = p0 + p1;
      rs += __shfl_xor(rs, 1);
      rs += __shfl_xor(rs, 2);
      rs += __shfl_xor(rs, 4);
      rs += __shfl_xor(rs, 8);
      l[r] = l[r] * corr + rs;
#pragma unroll
      for (int nt = 0; nt < 4; ++nt) oacc[nt][r] *= corr;
      pl[(half * 8 + r) * 32 + lr]      = (__bf16)p0;
      pl[(half * 8 + r) * 32 + 16 + lr] = (__bf16)p1;
    }
    __builtin_amdgcn_wave_barrier();
    // re-stripe P into A-fragment layout from wave-local LDS
    v8bf plo = *(const v8bf*)&pl[lr * 32 + half * 8];
    v8bf phi = *(const v8bf*)&pl[lr * 32 + 16 + half * 8];
    v16bf pa = SHUF16(plo, phi);
#pragma unroll
    for (int nt = 0; nt < 4; ++nt) {
      const __bf16* vp = vb + (size_t)(nt * 16 + lr) * cW + kv;
      v8bf vl = *(const v8bf*)(vp + half * 8);
      v8bf vh = *(const v8bf*)(vp + 16 + half * 8);
      oacc[nt] = WMMA_BF16(pa, SHUF16(vl, vh), oacc[nt]);
    }
    __builtin_amdgcn_wave_barrier();
  }
#pragma unroll
  for (int nt = 0; nt < 4; ++nt) {
    int d = nt * 16 + lr;
    int f = hh * cHD + d;
#pragma unroll
    for (int r = 0; r < 8; ++r) {
      int row = q0 + half * 8 + r;
      float val = oacc[nt][r] / l[r];
      O[((size_t)bc * cW + row) * cF + f] = (__bf16)val;   // w-major
    }
  }
}

// --------------------------- FFN (EXP = 4 bottleneck) ------------------------
__global__ void k_ffn1(const float* __restrict__ Z, const float* __restrict__ W3,
                       float* __restrict__ Z1) {
  size_t idx = (size_t)blockIdx.x * blockDim.x + threadIdx.x; // B*C*EXP*W
  int w = idx % cW;
  int e = (int)((idx / cW) % cEXP);
  size_t bc = idx / ((size_t)cEXP * cW);
  int c = (int)(bc % cC);
  const float* lp = Z + bc * (size_t)cF * cW + w;
  const float* wp = W3 + ((size_t)c * cEXP + e) * cF;
  float s = 0.f;
  for (int i = 0; i < cF; ++i) s += wp[i] * lp[(size_t)i * cW];
  Z1[idx] = s;
}

__global__ void k_ffn2_add(const float* __restrict__ Z1, const float* __restrict__ W4,
                           float* __restrict__ H) {
  size_t idx = (size_t)blockIdx.x * blockDim.x + threadIdx.x; // NX
  int w = idx % cW;
  int f = (idx / cW) % cF;
  size_t bc = idx / ((size_t)cF * cW);
  int c = (int)(bc % cC);
  float s = 0.f;
#pragma unroll
  for (int e = 0; e < cEXP; ++e) {
    float z = sqrelu(Z1[(bc * cEXP + e) * cW + w]);
    s += W4[((size_t)c * cF + f) * cEXP + e] * z;
  }
  H[idx] += s;
}

// --------------------------- finalize outputs --------------------------------
__global__ void k_final(const float* __restrict__ Xin, const float* __restrict__ H,
                        float* __restrict__ Out) {
  size_t idx = (size_t)blockIdx.x * blockDim.x + threadIdx.x; // NX
  size_t per_b = (size_t)cC * cF * cW;
  size_t b = idx / per_b;
  size_t rem = idx - b * per_b;
  Out[b * 2 * per_b + rem]          = Xin[idx];   // concat part 0: x
  Out[b * 2 * per_b + per_b + rem]  = H[idx];     // concat part 1: h
  Out[(size_t)cB * 2 * per_b + idx] = H[idx];     // second output: h
}

// ============================================================================
extern "C" void kernel_launch(void* const* d_in, const int* in_sizes, int n_in,
                              void* d_out, int out_size, void* d_ws, size_t ws_size,
                              hipStream_t stream) {
  const float* x     = (const float*)d_in[0];
  const float* n1g   = (const float*)d_in[1];
  const float* n1b   = (const float*)d_in[2];
  const float* n2g   = (const float*)d_in[3];
  const float* n2b   = (const float*)d_in[4];
  const float* n3g   = (const float*)d_in[5];
  const float* n3b   = (const float*)d_in[6];
  const float* n4g   = (const float*)d_in[7];
  const float* n4b   = (const float*)d_in[8];
  const float* n5g   = (const float*)d_in[9];
  const float* n5b   = (const float*)d_in[10];
  const float* glu_w = (const float*)d_in[11];
  const float* c1a_w = (const float*)d_in[12];
  const float* c1b_w = (const float*)d_in[13];
  const float* c1b_b = (const float*)d_in[14];
  const float* c2_w  = (const float*)d_in[15];
  const float* c2_b  = (const float*)d_in[16];
  const float* qw    = (const float*)d_in[17];
  const float* qcw   = (const float*)d_in[18];
  const float* qcb   = (const float*)d_in[19];
  const float* kw    = (const float*)d_in[20];
  const float* kcw   = (const float*)d_in[21];
  const float* kcb   = (const float*)d_in[22];
  const float* vw    = (const float*)d_in[23];
  const float* vcw   = (const float*)d_in[24];
  const float* vcb   = (const float*)d_in[25];
  const float* ow    = (const float*)d_in[26];
  const float* c3_w  = (const float*)d_in[27];
  const float* c4_w  = (const float*)d_in[28];

  // ----- workspace carving -----
  char* p = (char*)d_ws;
  auto carve = [&](size_t bytes) -> char* {
    char* r = p;
    p += (bytes + 255) & ~(size_t)255;
    return r;
  };
  float*  h    = (float*)carve((size_t)NX * 4);
  float*  t0   = (float*)carve((size_t)NX * 8);      // 2F-wide GLU out; later q/k mm
  float*  t1   = (float*)carve((size_t)NX * 4);      // c1a out; later v mm
  float*  lnf  = (float*)carve((size_t)NX * 4);
  __bf16* lnb  = (__bf16*)carve((size_t)NX * 2);     // LN bf16 (w-major); later attn out
  __bf16* qbuf = (__bf16*)carve((size_t)NX * 2);
  __bf16* kbuf = (__bf16*)carve((size_t)NX * 2);
  __bf16* vbuf = (__bf16*)carve((size_t)NX * 2);
  __bf16* wglu = (__bf16*)carve((size_t)cC * 2 * cF * cF * 2);
  __bf16* wc1a = (__bf16*)carve((size_t)cC * cF * cF * 2);
  __bf16* wq   = (__bf16*)carve((size_t)cC * cF * cF * 2);
  __bf16* wk   = (__bf16*)carve((size_t)cC * cF * cF * 2);
  __bf16* wv   = (__bf16*)carve((size_t)cC * cF * cF * 2);
  __bf16* wo   = (__bf16*)carve((size_t)cC * cF * cF * 2);
  float*  z1   = (float*)carve((size_t)cB * cC * cEXP * cW * 4);

  dim3 blk(256);
  auto cvt = [&](const float* src, __bf16* dst, int n) {
    k_cvt_bf16<<<dim3((n + 255) / 256), blk, 0, stream>>>(src, dst, n);
  };
  cvt(glu_w, wglu, cC * 2 * cF * cF);
  cvt(c1a_w, wc1a, cC * cF * cF);
  cvt(qw, wq, cC * cF * cF);
  cvt(kw, wk, cC * cF * cF);
  cvt(vw, wv, cC * cF * cF);
  cvt(ow, wo, cC * cF * cF);

  dim3 lnGrid(cB * cC, cW / 16);
  dim3 gemmBig(GN / 128, (2 * cF) / 64, cB * cC);   // M = 1024
  dim3 gemmStd(GN / 128, cF / 64, cB * cC);         // M = 512
  int nxBlocks = NX / 256;
  int pairBlocks = (NX / 2) / 256;

  // --- GLU branch: h = x + glu(mcl(ln1(x))) ---
  k_layernorm<<<lnGrid, blk, 0, stream>>>(x, n1g, n1b, lnf, lnb);
  k_gemm<<<gemmBig, blk, 0, stream>>>(wglu, lnb, t0, 2 * cF, 0);
  k_glu_add<<<nxBlocks, blk, 0, stream>>>(x, t0, h);

  // --- conv branch 1: h += sqrelu(mcl) + sqrelu(dwconv3) ---
  k_layernorm<<<lnGrid, blk, 0, stream>>>(h, n2g, n2b, lnf, lnb);
  k_gemm<<<gemmStd, blk, 0, stream>>>(wc1a, lnb, t1, cF, 0);
  k_sqrelu_dw3_add<<<nxBlocks, blk, 0, stream>>>(t1, lnf, c1b_w, c1b_b, h);

  // --- conv branch 2: h += dwconv9(ln3(h)) ---
  k_layernorm<<<lnGrid, blk, 0, stream>>>(h, n3g, n3b, lnf, lnb);
  k_dw9_add<<<nxBlocks, blk, 0, stream>>>(lnf, c2_w, c2_b, h);

  // --- attention: h += oproj(attn(rope(conv(qkv(ln4(h)))))) ---
  k_layernorm<<<lnGrid, blk, 0, stream>>>(h, n4g, n4b, lnf, lnb);
  k_gemm<<<gemmStd, blk, 0, stream>>>(wq, lnb, t0, cF, 0);
  k_gemm<<<gemmStd, blk, 0, stream>>>(wk, lnb, t0 + NX, cF, 0);
  k_gemm<<<gemmStd, blk, 0, stream>>>(wv, lnb, t1, cF, 0);
  k_convframes<<<pairBlocks, blk, 0, stream>>>(t0,      qcw, qcb, qbuf, 1, 1);
  k_convframes<<<pairBlocks, blk, 0, stream>>>(t0 + NX, kcw, kcb, kbuf, 1, 1);
  k_convframes<<<pairBlocks, blk, 0, stream>>>(t1,      vcw, vcb, vbuf, 0, 0);
  k_flash<<<dim3(cW / 128, cNH, cB * cC), blk, 0, stream>>>(qbuf, kbuf, vbuf, lnb);
  k_gemm<<<gemmStd, blk, 0, stream>>>(wo, lnb, h, cF, 1);  // accumulate into h

  // --- FFN: h += c4(sqrelu(c3(ln5(h)))) ---
  k_layernorm<<<lnGrid, blk, 0, stream>>>(h, n5g, n5b, lnf, lnb);
  k_ffn1<<<(cB * cC * cEXP * cW) / 256, blk, 0, stream>>>(lnf, c3_w, z1);
  k_ffn2_add<<<nxBlocks, blk, 0, stream>>>(z1, c4_w, h);

  // --- outputs: (concat(x, h, axis=1), h) ---
  k_final<<<nxBlocks, blk, 0, stream>>>(x, h, (float*)d_out);
}
